// AttentionNestedNERModel_53034256171370
// MI455X (gfx1250) — compile-verified, matrix operand order checked by reading, not run on hardware
//
#include <hip/hip_runtime.h>
#include <hip/hip_bf16.h>

// ---------------- problem dims ----------------
#define B_    64
#define S_    256
#define E_    200
#define H_    200
#define D2_   400
#define L_    4
#define C_    7
// padded K dims (multiples of 32 for wmma_f32_16x16x32_bf16)
#define KE_   224            // padded E
#define KENC_ 448            // [x(224) | h(224)] encoder fused K
#define NENC_ 800            // 4H
#define PH_   416            // padded D2 row pitch for encoder outputs / hw / ys
#define KDEC_ 1248           // [ctx(400)|h_t(400)|ctrl(4)|pad(28)|s(400)|pad(16)]
#define SOFF_ 832            // offset of s inside decoder xcat
#define NDEC_ 1600           // 4*D2
#define PYS_  416

typedef __attribute__((ext_vector_type(16))) __bf16 v16bf;
typedef __attribute__((ext_vector_type(8)))  float  v8f;
typedef __attribute__((ext_vector_type(8)))  __bf16 bf8v;
typedef __attribute__((ext_vector_type(2)))  __bf16 bf2v;

union FragU { v16bf v; unsigned u[8]; };
union Pk2U  { unsigned u; bf2v v; };

// Load a 16x32 bf16 WMMA fragment (A or B) from a row-major bf16 matrix.
// ISA layout (CDNA5 ISA): lanes 0-15 -> row = lane, K slots {0..7,16..23};
// lanes 16-31 -> row = lane-16, K slots {8..15,24..31}.
__device__ __forceinline__ v16bf load_frag(const __bf16* base, int pitch, int row,
                                           int k0, int lane) {
  FragU f;
  const unsigned* p =
      (const unsigned*)(base + (size_t)row * pitch + k0 + ((lane >> 4) << 3));
  f.u[0] = p[0]; f.u[1] = p[1]; f.u[2] = p[2];  f.u[3] = p[3];
  f.u[4] = p[8]; f.u[5] = p[9]; f.u[6] = p[10]; f.u[7] = p[11];
  return f.v;
}

__device__ __forceinline__ v8f wmma_bf(v16bf a, v16bf b, v8f c) {
  return __builtin_amdgcn_wmma_f32_16x16x32_bf16(false, a, false, b, (short)0, c,
                                                 false, false);
}

__device__ __forceinline__ float sigf(float x) { return 1.f / (1.f + __expf(-x)); }

// CDNA5 async global->LDS fill (tracked by ASYNCcnt). Inline asm so it works
// on both toolchains regardless of builtin arity. lds_off = low 32 bits of the
// flat shared address (== LDS byte address per the aperture layout).
__device__ __forceinline__ void async_g2l_b128(unsigned lds_off, const __bf16* gsrc) {
  asm volatile("global_load_async_to_lds_b128 %0, %1, off"
               :: "v"(lds_off), "v"(gsrc) : "memory");
}
__device__ __forceinline__ void wait_async0() {
  asm volatile("s_wait_asynccnt 0x0" ::: "memory");
}

// ---------------- weight preparation (fp32 -> padded bf16) ----------------
__global__ void k_prep_encw(const float* Wi_f, const float* Wh_f,
                            const float* Wi_b, const float* Wh_b, __bf16* Wenc) {
  int idx = blockIdx.x * blockDim.x + threadIdx.x;          // 2*800*448
  if (idx >= 2 * NENC_ * KENC_) return;
  int dir = idx / (NENC_ * KENC_);
  int rem = idx % (NENC_ * KENC_);
  int n = rem / KENC_, k = rem % KENC_;
  const float* Wi = dir ? Wi_b : Wi_f;
  const float* Wh = dir ? Wh_b : Wh_f;
  float v = 0.f;
  if (k < E_) v = Wi[n * E_ + k];
  else if (k >= KE_ && k < KE_ + H_) v = Wh[n * H_ + (k - KE_)];
  Wenc[idx] = (__bf16)v;
}

__global__ void k_prep_decw(const float* Wi_d, const float* Wh_d, __bf16* Wdec) {
  int idx = blockIdx.x * blockDim.x + threadIdx.x;          // 1600*1248
  if (idx >= NDEC_ * KDEC_) return;
  int n = idx / KDEC_, k = idx % KDEC_;
  float v = 0.f;
  if (k < 2 * D2_ + L_) v = Wi_d[n * (2 * D2_ + L_) + k];
  else if (k >= SOFF_ && k < SOFF_ + D2_) v = Wh_d[n * D2_ + (k - SOFF_)];
  Wdec[idx] = (__bf16)v;
}

__global__ void k_prep_small(const float* bi_f, const float* bh_f,
                             const float* bi_b, const float* bh_b,
                             const float* bi_d, const float* bh_d,
                             const float* W_att, const float* W2,
                             float* benc, float* bdec, __bf16* Watt, __bf16* W2b) {
  int idx = blockIdx.x * blockDim.x + threadIdx.x;
  if (idx < 2 * NENC_) {
    int dir = idx / NENC_, j = idx % NENC_;
    benc[idx] = dir ? (bi_b[j] + bh_b[j]) : (bi_f[j] + bh_f[j]);
    return;
  }
  idx -= 2 * NENC_;
  if (idx < NDEC_) { bdec[idx] = bi_d[idx] + bh_d[idx]; return; }
  idx -= NDEC_;
  if (idx < D2_ * PH_) {
    int n = idx / PH_, k = idx % PH_;
    Watt[idx] = (__bf16)((k < D2_) ? W_att[n * D2_ + k] : 0.f);
    return;
  }
  idx -= D2_ * PH_;
  if (idx < 16 * PH_) {
    int n = idx / PH_, k = idx % PH_;
    W2b[idx] = (__bf16)((n < C_ && k < D2_) ? W2[n * D2_ + k] : 0.f);
  }
}

__global__ void k_zero_pads(__bf16* henc, __bf16* ys) {
  int idx = blockIdx.x * blockDim.x + threadIdx.x;
  if (idx < S_ * B_ * 16) {
    henc[(size_t)(idx / 16) * PH_ + D2_ + (idx % 16)] = (__bf16)0.f;
    return;
  }
  idx -= S_ * B_ * 16;
  if (idx < L_ * S_ * B_ * 16) {
    ys[(size_t)(idx / 16) * PYS_ + D2_ + (idx % 16)] = (__bf16)0.f;
  }
}

// ---------------- embedding gather -> padded bf16 [S][B][KE] ----------------
__global__ void k_embed(const int* seqs, const float* emb, __bf16* xbf) {
  int idx = blockIdx.x * blockDim.x + threadIdx.x;          // S*B*KE
  if (idx >= S_ * B_ * KE_) return;
  int k = idx % KE_, sb = idx / KE_;
  int b = sb % B_, s = sb / B_;
  int tok = seqs[b * S_ + s];
  float v = (k < E_) ? emb[(size_t)tok * E_ + k] : 0.f;
  xbf[idx] = (__bf16)v;
}

// ---------------- bidirectional LSTM encoder ----------------
__global__ void __launch_bounds__(512)
k_encoder(const __bf16* xbf, const __bf16* Wenc, const float* benc,
          __bf16* henc, float* gws) {
  int dir = blockIdx.x;
  const __bf16* W   = Wenc + (size_t)dir * NENC_ * KENC_;
  const float*  bia = benc + dir * NENC_;
  float*        g   = gws  + (size_t)dir * B_ * NENC_;

  __shared__ __align__(16) __bf16 xcat[B_][KENC_];          // 57344 B
  int tid = threadIdx.x, lane = tid & 31, wave = tid >> 5;
  unsigned ldsbase = (unsigned)(size_t)(&xcat[0][0]);

  for (int i = tid; i < B_ * KENC_; i += 512) {             // zero h region + pads
    int k = i % KENC_;
    if (k >= KE_) ((__bf16*)xcat)[i] = (__bf16)0.f;
  }
  float creg[25];
#pragma unroll
  for (int r = 0; r < 25; ++r) creg[r] = 0.f;
  __syncthreads();

  for (int t = 0; t < S_; ++t) {
    int s = dir ? (S_ - 1 - t) : t;
    // async-stage x_t: 64*224 bf16 = 1792 x 16B chunks via ASYNC path
    for (int i = 0; i < 4; ++i) {
      int c = tid + 512 * i;
      if (c < (B_ * KE_) / 8) {
        int b = c / (KE_ / 8), kk = c % (KE_ / 8);
        unsigned dst = ldsbase + (unsigned)((b * KENC_ + kk * 8) * 2);
        async_g2l_b128(dst, xbf + (size_t)s * B_ * KE_ + c * 8);
      }
    }
    wait_async0();
    __syncthreads();

    for (int i = 0; i < 13; ++i) {                          // 200 tiles / 16 waves
      int tile = wave + 16 * i;
      if (tile < 200) {
        int m0 = (tile & 3) * 16, n0 = (tile >> 2) * 16;
        v8f acc = {0.f, 0.f, 0.f, 0.f, 0.f, 0.f, 0.f, 0.f};
#pragma unroll 2
        for (int kt = 0; kt < KENC_ / 32; ++kt) {
          v16bf a  = load_frag(&xcat[0][0], KENC_, m0 + (lane & 15), kt * 32, lane);
          v16bf bb = load_frag(W,           KENC_, n0 + (lane & 15), kt * 32, lane);
          acc = wmma_bf(a, bb, acc);
        }
        int n = n0 + (lane & 15), mb = m0 + ((lane >> 4) << 3);
#pragma unroll
        for (int r = 0; r < 8; ++r) g[(mb + r) * NENC_ + n] = acc[r];
      }
    }
    __threadfence();
    __syncthreads();

#pragma unroll
    for (int r = 0; r < 25; ++r) {                          // gates (512*25 = 64*200)
      int flat = tid + 512 * r;
      int b = flat / H_, j = flat % H_;
      float gi = g[b * NENC_ + j]          + bia[j];
      float gf = g[b * NENC_ + H_ + j]     + bia[H_ + j];
      float gg = g[b * NENC_ + 2 * H_ + j] + bia[2 * H_ + j];
      float go = g[b * NENC_ + 3 * H_ + j] + bia[3 * H_ + j];
      float c = sigf(gf) * creg[r] + sigf(gi) * tanhf(gg);
      creg[r] = c;
      float h = sigf(go) * tanhf(c);
      xcat[b][KE_ + j] = (__bf16)h;                         // feed back
      henc[(size_t)(s * B_ + b) * PH_ + dir * H_ + j] = (__bf16)h;
    }
    __threadfence();
    __syncthreads();
  }
}

// ---------------- attention projection: hw = h @ W_att^T ----------------
__global__ void __launch_bounds__(256)
k_gemm_att(const __bf16* henc, const __bf16* Watt, __bf16* hw) {
  int gw = (blockIdx.x * 256 + threadIdx.x) >> 5;
  int lane = threadIdx.x & 31;
  if (gw >= (S_ * B_ / 16) * (D2_ / 16)) return;
  int mt = gw / (D2_ / 16), nt = gw % (D2_ / 16);
  int m0 = mt * 16, n0 = nt * 16;
  v8f acc = {0.f, 0.f, 0.f, 0.f, 0.f, 0.f, 0.f, 0.f};
  for (int kt = 0; kt < PH_ / 32; ++kt) {
    v16bf a  = load_frag(henc, PH_, m0 + (lane & 15), kt * 32, lane);
    v16bf bb = load_frag(Watt, PH_, n0 + (lane & 15), kt * 32, lane);
    acc = wmma_bf(a, bb, acc);
  }
  int n = n0 + (lane & 15), mb = m0 + ((lane >> 4) << 3);
#pragma unroll
  for (int r = 0; r < 8; ++r) hw[(size_t)(mb + r) * PH_ + n] = (__bf16)acc[r];
}

// ---------------- attention decoder (persistent single workgroup) ----------------
__global__ void __launch_bounds__(1024)
k_decoder(const __bf16* henc, const __bf16* hw, const __bf16* Wdec,
          const float* bdec, float* g, float* scores, __bf16* ys) {
  __shared__ __align__(16) __bf16 xcat[B_][KDEC_];          // 159744 B
  __shared__ float red[2 * B_];
  int tid = threadIdx.x, lane = tid & 31, wave = tid >> 5;
  unsigned ldsbase = (unsigned)(size_t)(&xcat[0][0]);

  for (int i = tid; i < B_ * KDEC_; i += 1024)              // s=0, pads=0
    ((__bf16*)xcat)[i] = (__bf16)0.f;
  float creg[25];
#pragma unroll
  for (int r = 0; r < 25; ++r) creg[r] = 0.f;
  __syncthreads();

  for (int lvl = 0; lvl < L_; ++lvl) {
    for (int i = tid; i < B_ * L_; i += 1024) {             // one-hot ctrl
      int b = i / L_, l = i % L_;
      xcat[b][2 * D2_ + l] = (__bf16)((l == lvl) ? 1.f : 0.f);
    }
    __syncthreads();

    for (int step = 0; step < S_; ++step) {
      // --- 0) async-stage h_t into xcat[:,400..800): 3200 x 16B chunks ---
      for (int i = 0; i < 4; ++i) {
        int c = tid + 1024 * i;
        if (c < (B_ * D2_) / 8) {
          int b = c / (D2_ / 8), kk = c % (D2_ / 8);
          unsigned dst = ldsbase + (unsigned)((b * KDEC_ + D2_ + kk * 8) * 2);
          async_g2l_b128(dst, henc + (size_t)(step * B_ + b) * PH_ + kk * 8);
        }
      }

      // --- 1) attention scores (overlaps with async h_t fill) ---
#pragma unroll 1
      for (int r = 0; r < 16; ++r) {
        int flat = tid + 1024 * r;                          // = si*64 + b
        int b = flat & 63;
        const bf8v* hwv = (const bf8v*)(hw + (size_t)flat * PH_);
        const bf8v* sv  = (const bf8v*)&xcat[b][SOFF_];
        if (r < 15) __builtin_prefetch(hw + (size_t)(flat + 1024) * PH_, 0, 1);
        float acc = 0.f;
#pragma unroll 2
        for (int q = 0; q < D2_ / 8; ++q) {
          bf8v hv = hwv[q], s8 = sv[q];
#pragma unroll
          for (int e = 0; e < 8; ++e) acc += (float)hv[e] * (float)s8[e];
        }
        scores[flat] = acc;
      }
      __threadfence();
      __syncthreads();

      // --- 2) softmax stats: 16 lanes per batch column + shfl_xor tree ---
      {
        int b = tid >> 4, t16 = tid & 15;
        float m = -1e30f;
#pragma unroll
        for (int i = 0; i < 16; ++i)
          m = fmaxf(m, scores[(t16 + 16 * i) * B_ + b]);
        for (int mk = 1; mk < 16; mk <<= 1) m = fmaxf(m, __shfl_xor(m, mk, 32));
        float sum = 0.f;
#pragma unroll
        for (int i = 0; i < 16; ++i)
          sum += __expf(scores[(t16 + 16 * i) * B_ + b] - m);
        for (int mk = 1; mk < 16; mk <<= 1) sum += __shfl_xor(sum, mk, 32);
        if (t16 == 0) { red[b] = m; red[B_ + b] = 1.f / sum; }
      }
      __syncthreads();

      // --- 2b) materialize softmax weights in-place ---
#pragma unroll 1
      for (int r = 0; r < 16; ++r) {
        int flat = tid + 1024 * r;
        int b = flat & 63;
        scores[flat] = __expf(scores[flat] - red[b]) * red[B_ + b];
      }
      __threadfence();
      __syncthreads();

      // --- 3) context: paired dword loads, ctx -> xcat[:,0..400) ---
#pragma unroll 1
      for (int r = 0; r < 13; ++r) {
        int p = tid + 1024 * r;                             // pair index < 12800
        if (p < (B_ * D2_) / 2) {
          int b = p / (D2_ / 2), jp = p % (D2_ / 2);
          float a0 = 0.f, a1 = 0.f;
          for (int si = 0; si < S_; ++si) {
            float wgt = scores[si * B_ + b];
            Pk2U pk;
            pk.u = *(const unsigned*)(henc + (size_t)(si * B_ + b) * PH_ + 2 * jp);
            a0 += wgt * (float)pk.v[0];
            a1 += wgt * (float)pk.v[1];
          }
          bf2v o; o[0] = (__bf16)a0; o[1] = (__bf16)a1;
          *(bf2v*)&xcat[b][2 * jp] = o;
        }
      }
      wait_async0();                                        // h_t fill landed
      __syncthreads();

      // --- 4) fused decoder GEMM (400 tiles / 32 waves, K=39 wmma) ---
      for (int i = 0; i < 13; ++i) {
        int tile = wave + 32 * i;
        if (tile < 400) {
          int m0 = (tile & 3) * 16, n0 = (tile >> 2) * 16;
          v8f acc = {0.f, 0.f, 0.f, 0.f, 0.f, 0.f, 0.f, 0.f};
#pragma unroll 1
          for (int kt = 0; kt < KDEC_ / 32; ++kt) {
            v16bf a  = load_frag(&xcat[0][0], KDEC_, m0 + (lane & 15), kt * 32, lane);
            v16bf bb = load_frag(Wdec,        KDEC_, n0 + (lane & 15), kt * 32, lane);
            acc = wmma_bf(a, bb, acc);
          }
          int n = n0 + (lane & 15), mb = m0 + ((lane >> 4) << 3);
#pragma unroll
          for (int r = 0; r < 8; ++r) g[(mb + r) * NDEC_ + n] = acc[r];
        }
      }
      __threadfence();
      __syncthreads();

      // --- 5) LSTM cell update (fp32 c in registers) ---
#pragma unroll
      for (int r = 0; r < 25; ++r) {
        int flat = tid + 1024 * r;
        int b = flat / D2_, j = flat % D2_;
        float gi = g[b * NDEC_ + j]           + bdec[j];
        float gf = g[b * NDEC_ + D2_ + j]     + bdec[D2_ + j];
        float gg = g[b * NDEC_ + 2 * D2_ + j] + bdec[2 * D2_ + j];
        float go = g[b * NDEC_ + 3 * D2_ + j] + bdec[3 * D2_ + j];
        float c = sigf(gf) * creg[r] + sigf(gi) * tanhf(gg);
        creg[r] = c;
        float sv = sigf(go) * tanhf(c);
        xcat[b][SOFF_ + j] = (__bf16)sv;                    // recurrent feedback
        ys[((size_t)(lvl * S_ + step) * B_ + b) * PYS_ + j] = (__bf16)sv;
      }
      __threadfence();
      __syncthreads();
    }
  }
}

// ---------------- final projection: logits = ys @ W2^T + b2 ----------------
__global__ void __launch_bounds__(256)
k_logits(const __bf16* ys, const __bf16* W2b, const float* b2, float* out) {
  int gw = (blockIdx.x * 256 + threadIdx.x) >> 5;
  int lane = threadIdx.x & 31;
  if (gw >= (L_ * S_ * B_) / 16) return;
  int m0 = gw * 16;
  v8f acc = {0.f, 0.f, 0.f, 0.f, 0.f, 0.f, 0.f, 0.f};
  for (int kt = 0; kt < PYS_ / 32; ++kt) {
    v16bf a  = load_frag(ys,  PYS_, m0 + (lane & 15), kt * 32, lane);
    v16bf bb = load_frag(W2b, PYS_, (lane & 15),      kt * 32, lane);
    acc = wmma_bf(a, bb, acc);
  }
  int n = lane & 15, mb = m0 + ((lane >> 4) << 3);
  if (n < C_) {
#pragma unroll
    for (int r = 0; r < 8; ++r)
      out[(size_t)(mb + r) * C_ + n] = acc[r] + b2[n];
  }
}

// ---------------- host launch ----------------
extern "C" void kernel_launch(void* const* d_in, const int* in_sizes, int n_in,
                              void* d_out, int out_size, void* d_ws, size_t ws_size,
                              hipStream_t stream) {
  (void)in_sizes; (void)n_in; (void)out_size; (void)ws_size;
  const int*   seqs = (const int*)  d_in[0];
  const float* emb  = (const float*)d_in[2];
  const float* Wi_f = (const float*)d_in[3];
  const float* Wh_f = (const float*)d_in[4];
  const float* bi_f = (const float*)d_in[5];
  const float* bh_f = (const float*)d_in[6];
  const float* Wi_b = (const float*)d_in[7];
  const float* Wh_b = (const float*)d_in[8];
  const float* bi_b = (const float*)d_in[9];
  const float* bh_b = (const float*)d_in[10];
  const float* Watt = (const float*)d_in[11];
  const float* Wi_d = (const float*)d_in[12];
  const float* Wh_d = (const float*)d_in[13];
  const float* bi_d = (const float*)d_in[14];
  const float* bh_d = (const float*)d_in[15];
  const float* W2   = (const float*)d_in[16];
  const float* b2   = (const float*)d_in[17];
  float* out = (float*)d_out;

  char* w = (char*)d_ws;
  auto carve = [&](size_t bytes) {
    char* p = w;
    w += (bytes + 255) & ~(size_t)255;
    return p;
  };
  __bf16* WENC = (__bf16*)carve((size_t)2 * NENC_ * KENC_ * 2);
  float*  BENC = (float*) carve((size_t)2 * NENC_ * 4);
  __bf16* WDEC = (__bf16*)carve((size_t)NDEC_ * KDEC_ * 2);
  float*  BDEC = (float*) carve((size_t)NDEC_ * 4);
  __bf16* WATT = (__bf16*)carve((size_t)D2_ * PH_ * 2);
  __bf16* W2B  = (__bf16*)carve((size_t)16 * PH_ * 2);
  __bf16* XBF  = (__bf16*)carve((size_t)S_ * B_ * KE_ * 2);
  __bf16* HENC = (__bf16*)carve((size_t)S_ * B_ * PH_ * 2);
  __bf16* HW   = (__bf16*)carve((size_t)S_ * B_ * PH_ * 2);
  float*  GENC = (float*) carve((size_t)2 * B_ * NENC_ * 4);
  float*  GDEC = (float*) carve((size_t)B_ * NDEC_ * 4);
  float*  SCOR = (float*) carve((size_t)S_ * B_ * 4);
  __bf16* YS   = (__bf16*)carve((size_t)L_ * S_ * B_ * PYS_ * 2);

  {
    int n = 2 * NENC_ * KENC_;
    k_prep_encw<<<(n + 255) / 256, 256, 0, stream>>>(Wi_f, Wh_f, Wi_b, Wh_b, WENC);
  }
  {
    int n = NDEC_ * KDEC_;
    k_prep_decw<<<(n + 255) / 256, 256, 0, stream>>>(Wi_d, Wh_d, WDEC);
  }
  {
    int n = 2 * NENC_ + NDEC_ + D2_ * PH_ + 16 * PH_;
    k_prep_small<<<(n + 255) / 256, 256, 0, stream>>>(
        bi_f, bh_f, bi_b, bh_b, bi_d, bh_d, Watt, W2, BENC, BDEC, WATT, W2B);
  }
  {
    int n = S_ * B_ * 16 + L_ * S_ * B_ * 16;
    k_zero_pads<<<(n + 255) / 256, 256, 0, stream>>>(HENC, YS);
  }
  {
    int n = S_ * B_ * KE_;
    k_embed<<<(n + 255) / 256, 256, 0, stream>>>(seqs, emb, XBF);
  }

  k_encoder<<<2, 512, 0, stream>>>(XBF, WENC, BENC, HENC, GENC);

  {
    int tiles = (S_ * B_ / 16) * (D2_ / 16);
    k_gemm_att<<<(tiles + 7) / 8, 256, 0, stream>>>(HENC, WATT, HW);
  }

  k_decoder<<<1, 1024, 0, stream>>>(HENC, HW, WDEC, BDEC, GDEC, SCOR, YS);

  {
    int tiles = (L_ * S_ * B_) / 16;
    k_logits<<<(tiles + 7) / 8, 256, 0, stream>>>(YS, W2B, b2, out);
  }
}
